// BoundaryConvLayer_74500502716662
// MI455X (gfx1250) — matrix operand order repaired and checked by reading, not compiled, not run on hardware
//
#include <hip/hip_runtime.h>

typedef __attribute__((ext_vector_type(16))) __bf16 v16bf;
typedef __attribute__((ext_vector_type(2)))  __bf16 v2bf;
typedef __attribute__((ext_vector_type(8)))  float  v8f;

#define DIM 128
#define EPSV 1e-4f
#define LNEPS 1e-5f

union Pack2 { unsigned u; v2bf v; };

// ---- f32 -> packed bf16x2: HW convert if declared, else 3-op VALU path ----
static __device__ __forceinline__ v2bf cvt2(float a, float b) {
#if __has_builtin(__builtin_amdgcn_cvt_pk_bf16_f32)
    return __builtin_amdgcn_cvt_pk_bf16_f32(a, b);
#else
    // round-half-up to bf16, then one v_perm_b32 extracts both high halves:
    // dst[15:0] = (bits(a)+0x8000)[31:16], dst[31:16] = (bits(b)+0x8000)[31:16]
    unsigned ua = __float_as_uint(a) + 0x8000u;
    unsigned ub = __float_as_uint(b) + 0x8000u;
    Pack2 r;
    r.u = __builtin_amdgcn_perm(ub, ua, 0x07060302u);
    return r.v;
#endif
}

// fast softplus: v_exp_f32 / v_log_f32 transcendentals, no cancellation
static __device__ __forceinline__ float softplus_fast(float v) {
    float e = __expf(-fabsf(v));
    return fmaxf(v, 0.f) + __logf(1.f + e);
}

union AFrag { v2bf p[8]; v16bf v; };

// ---------------------------------------------------------------------------
// Kernel 1: fused triple GEMM over FULL 16-row tiles only (no guards in the
// hot path).  h = xW_fc + b_fc ; rate = softplus(xW_rate)+eps ; gam = xW_rob
// + b_rob.  One wave per tile, bf16 WMMA, weights transposed bf16 in LDS.
// ---------------------------------------------------------------------------
__global__ __launch_bounds__(256) void fused_gemm_kernel(
    const float* __restrict__ x,
    const float* __restrict__ W_fc, const float* __restrict__ b_fc,
    const float* __restrict__ W_rate,
    const float* __restrict__ W_rob, const float* __restrict__ b_rob,
    float* __restrict__ h, float* __restrict__ rate, float* __restrict__ gam,
    int ntilesFull)
{
    // Weights transposed to [mat][n][k] so B fragments are contiguous in k.
    __shared__ __align__(32) __bf16 ldsW[3][DIM][DIM];   // 96 KB

    const float* Ws[3] = { W_fc, W_rate, W_rob };
    for (int mat = 0; mat < 3; ++mat) {
        const float* W = Ws[mat];
        // convert pairs along k (contiguous in LDS target) -> packed b32 stores
        for (int idx = threadIdx.x; idx < (DIM * DIM) / 2; idx += blockDim.x) {
            int n = idx & 127;            // column
            int k = (idx >> 7) * 2;       // even k
            v2bf pr = cvt2(W[(size_t)k * DIM + n], W[(size_t)(k + 1) * DIM + n]);
            *(v2bf*)&ldsW[mat][n][k] = pr;
        }
    }
    __syncthreads();

    const int lane        = threadIdx.x & 31;
    const int waveInBlk   = threadIdx.x >> 5;
    const int wavesPerBlk = blockDim.x >> 5;
    const int tileStride  = gridDim.x * wavesPerBlk;
    const int m     = lane & 15;                  // A-matrix row for this lane
    const int koff  = (lane < 16) ? 0 : 16;       // A layout: K sub-offset
    const int khalf = (lane < 16) ? 0 : 16;       // B layout: K half
    const int ncol  = lane & 15;                  // B/C/D column
    const int mbase = (lane < 16) ? 0 : 8;        // C/D row base

    for (int tile = blockIdx.x * wavesPerBlk + waveInBlk; tile < ntilesFull;
         tile += tileStride) {

        const int m0 = tile * 16;
        const float* xr = x + (size_t)(m0 + m) * DIM;

        // Prefetch this wave's next tile of x (global_prefetch_b8 path).
        if (tile + tileStride < ntilesFull)
            __builtin_prefetch(xr + (size_t)tileStride * 16 * DIM, 0, 1);

        // Build 4 A fragments (K = 0..127 in chunks of 32), contiguous k = 16*half + s (same as B).
        v16bf afr[4];
        #pragma unroll
        for (int kk = 0; kk < 4; ++kk) {
            const float* p = xr + kk * 32 + koff;
            float4 q0 = *(const float4*)(p);
            float4 q1 = *(const float4*)(p + 4);
            float4 q2 = *(const float4*)(p + 8);
            float4 q3 = *(const float4*)(p + 12);
            AFrag a;
            a.p[0] = cvt2(q0.x, q0.y);
            a.p[1] = cvt2(q0.z, q0.w);
            a.p[2] = cvt2(q1.x, q1.y);
            a.p[3] = cvt2(q1.z, q1.w);
            a.p[4] = cvt2(q2.x, q2.y);
            a.p[5] = cvt2(q2.z, q2.w);
            a.p[6] = cvt2(q3.x, q3.y);
            a.p[7] = cvt2(q3.z, q3.w);
            afr[kk] = a.v;
        }

        const size_t rowBase = (size_t)(m0 + mbase) * DIM;   // lane's first out row

        #pragma unroll
        for (int mat = 0; mat < 3; ++mat) {
            #pragma unroll
            for (int n0 = 0; n0 < DIM; n0 += 16) {
                v8f acc = {0.f, 0.f, 0.f, 0.f, 0.f, 0.f, 0.f, 0.f};
                #pragma unroll
                for (int kk = 0; kk < 4; ++kk) {
                    const v16bf bfr =
                        *(const v16bf*)&ldsW[mat][n0 + ncol][kk * 32 + khalf];
                    acc = __builtin_amdgcn_wmma_f32_16x16x32_bf16(
                        false, afr[kk], false, bfr, (short)0, acc, false, false);
                }
                const int n = n0 + ncol;

                // Compute all 8 epilogue values first (distinct regs), then a
                // store burst off one base address with immediate offsets.
                float vals[8];
                if (mat == 0) {
                    const float bias = b_fc[n];
                    #pragma unroll
                    for (int r = 0; r < 8; ++r) vals[r] = acc[r] + bias;
                    float* bp = h + rowBase + n;
                    #pragma unroll
                    for (int r = 0; r < 8; ++r) bp[(size_t)r * DIM] = vals[r];
                } else if (mat == 1) {
                    #pragma unroll
                    for (int r = 0; r < 8; ++r) vals[r] = softplus_fast(acc[r]) + EPSV;
                    float* bp = rate + rowBase + n;
                    #pragma unroll
                    for (int r = 0; r < 8; ++r) bp[(size_t)r * DIM] = vals[r];
                } else {
                    const float bias = b_rob[n];
                    #pragma unroll
                    for (int r = 0; r < 8; ++r) vals[r] = acc[r] + bias;
                    float* bp = gam + rowBase + n;
                    #pragma unroll
                    for (int r = 0; r < 8; ++r) bp[(size_t)r * DIM] = vals[r];
                }
            }
        }
    }
}

// ---------------------------------------------------------------------------
// Tail kernel: scalar path for the (rare) N % 16 remainder rows.
// ---------------------------------------------------------------------------
__global__ void tail_gemm_kernel(
    const float* __restrict__ x,
    const float* __restrict__ W_fc, const float* __restrict__ b_fc,
    const float* __restrict__ W_rate,
    const float* __restrict__ W_rob, const float* __restrict__ b_rob,
    float* __restrict__ h, float* __restrict__ rate, float* __restrict__ gam,
    int rowStart, int N)
{
    int total = (N - rowStart) * DIM;
    for (int idx = blockIdx.x * blockDim.x + threadIdx.x; idx < total;
         idx += gridDim.x * blockDim.x) {
        int r = rowStart + idx / DIM;
        int n = idx % DIM;
        const float* xr = x + (size_t)r * DIM;
        float s0 = 0.f, s1 = 0.f, s2 = 0.f;
        for (int k = 0; k < DIM; ++k) {
            float xv = xr[k];
            s0 = fmaf(xv, W_fc[(size_t)k * DIM + n], s0);
            s1 = fmaf(xv, W_rate[(size_t)k * DIM + n], s1);
            s2 = fmaf(xv, W_rob[(size_t)k * DIM + n], s2);
        }
        h[(size_t)r * DIM + n]    = s0 + b_fc[n];
        rate[(size_t)r * DIM + n] = softplus_fast(s1) + EPSV;
        gam[(size_t)r * DIM + n]  = s2 + b_rob[n];
    }
}

// ---------------------------------------------------------------------------
// Kernel 0: zero the scatter accumulators (partial [N*128] f32, cnt [N] int).
// ---------------------------------------------------------------------------
__global__ void zero_ws_kernel(float* __restrict__ partial, int* __restrict__ cnt,
                               size_t nPartial, int N)
{
    size_t i = (size_t)blockIdx.x * blockDim.x + threadIdx.x;
    size_t stride = (size_t)gridDim.x * blockDim.x;
    for (size_t j = i; j < nPartial; j += stride) partial[j] = 0.f;
    for (size_t j = i; j < (size_t)N;  j += stride) cnt[j] = 0;
}

// ---------------------------------------------------------------------------
// Kernel 2: edge scatter.  agg[row] = cnt(row)*h[row] + sum h[col]; we only
// atomically accumulate sum h[col] and the per-row count (halves the traffic).
// One wave per edge; float4 gather (hits L2: h is 51 MB < 192 MB).
// ---------------------------------------------------------------------------
__global__ __launch_bounds__(256) void edge_scatter_kernel(
    const int* __restrict__ ei, const float* __restrict__ h,
    float* __restrict__ partial, int* __restrict__ cnt, long long E)
{
    const int lane = threadIdx.x & 31;
    long long wid = ((long long)blockIdx.x * blockDim.x + threadIdx.x) >> 5;
    long long nw  = ((long long)gridDim.x * blockDim.x) >> 5;
    for (long long e = wid; e < E; e += nw) {
        const int row = ei[e];
        const int col = ei[E + e];
        if (lane == 0) atomicAdd(&cnt[row], 1);
        const float4 hv = *(const float4*)&h[(size_t)col * DIM + lane * 4];
        float* dst = &partial[(size_t)row * DIM + lane * 4];
        atomicAdd(dst + 0, hv.x);
        atomicAdd(dst + 1, hv.y);
        atomicAdd(dst + 2, hv.z);
        atomicAdd(dst + 3, hv.w);
    }
}

// ---------------------------------------------------------------------------
// Kernel 3: combine + LayerNorm.  One wave per node (32 lanes x 4 dims).
// ---------------------------------------------------------------------------
__global__ __launch_bounds__(256) void finalize_kernel(
    const float* __restrict__ h, const float* __restrict__ rate,
    const float* __restrict__ gam, const float* __restrict__ partial,
    const int* __restrict__ cnt, const int* __restrict__ degree,
    const float* __restrict__ ln_g, const float* __restrict__ ln_b,
    float* __restrict__ out, int N)
{
    const int lane = threadIdx.x & 31;
    int wid = (int)(((long long)blockIdx.x * blockDim.x + threadIdx.x) >> 5);
    int nw  = (int)(((long long)gridDim.x * blockDim.x) >> 5);
    const float4 lg = *(const float4*)&ln_g[lane * 4];
    const float4 lb = *(const float4*)&ln_b[lane * 4];

    for (int i = wid; i < N; i += nw) {
        const size_t base = (size_t)i * DIM + lane * 4;
        const float4 hv = *(const float4*)&h[base];
        const float4 pv = *(const float4*)&partial[base];
        const float4 rv = *(const float4*)&rate[base];
        const float4 gv = *(const float4*)&gam[base];
        const float c  = (float)cnt[i];
        const float dg = (float)degree[i];

        float y0 = __fdividef(rv.x * (pv.x + c * hv.x) + gv.x, 1.0f + rv.x * dg + EPSV);
        float y1 = __fdividef(rv.y * (pv.y + c * hv.y) + gv.y, 1.0f + rv.y * dg + EPSV);
        float y2 = __fdividef(rv.z * (pv.z + c * hv.z) + gv.z, 1.0f + rv.z * dg + EPSV);
        float y3 = __fdividef(rv.w * (pv.w + c * hv.w) + gv.w, 1.0f + rv.w * dg + EPSV);

        float s  = y0 + y1 + y2 + y3;
        float sq = y0 * y0 + y1 * y1 + y2 * y2 + y3 * y3;
        #pragma unroll
        for (int off = 16; off > 0; off >>= 1) {
            s  += __shfl_xor(s,  off, 32);
            sq += __shfl_xor(sq, off, 32);
        }
        const float mean = s * (1.0f / DIM);
        const float var  = sq * (1.0f / DIM) - mean * mean;
        const float inv  = rsqrtf(var + LNEPS);

        float4 o;
        o.x = (y0 - mean) * inv * lg.x + lb.x;
        o.y = (y1 - mean) * inv * lg.y + lb.y;
        o.z = (y2 - mean) * inv * lg.z + lb.z;
        o.w = (y3 - mean) * inv * lg.w + lb.w;
        *(float4*)&out[base] = o;
    }
}

// ---------------------------------------------------------------------------
extern "C" void kernel_launch(void* const* d_in, const int* in_sizes, int n_in,
                              void* d_out, int out_size, void* d_ws, size_t ws_size,
                              hipStream_t stream)
{
    const float* x      = (const float*)d_in[0];
    const int*   ei     = (const int*)  d_in[1];
    const int*   degree = (const int*)  d_in[2];
    const float* W_fc   = (const float*)d_in[3];
    const float* b_fc   = (const float*)d_in[4];
    const float* W_rate = (const float*)d_in[5];
    const float* W_rob  = (const float*)d_in[6];
    const float* b_rob  = (const float*)d_in[7];
    const float* ln_g   = (const float*)d_in[8];
    const float* ln_b   = (const float*)d_in[9];
    float* out = (float*)d_out;

    const int       N = in_sizes[0] / DIM;
    const long long E = (long long)in_sizes[1] / 2;

    // Workspace layout: h | rate | gam | partial (each N*128 f32), then cnt (N int)
    float* h       = (float*)d_ws;
    float* rate    = h    + (size_t)N * DIM;
    float* gam     = rate + (size_t)N * DIM;
    float* partial = gam  + (size_t)N * DIM;
    int*   cnt     = (int*)(partial + (size_t)N * DIM);

    const int ntilesFull = N / 16;
    const int tailRows   = N - ntilesFull * 16;

    // 0) zero accumulators
    {
        int blocks = 2048;
        zero_ws_kernel<<<blocks, 256, 0, stream>>>(partial, cnt, (size_t)N * DIM, N);
    }
    // 1) fused triple GEMM (WMMA bf16) over full tiles
    if (ntilesFull > 0) {
        int blocks = (ntilesFull + 7) / 8;        // 8 waves per 256-thread block
        fused_gemm_kernel<<<blocks, 256, 0, stream>>>(
            x, W_fc, b_fc, W_rate, W_rob, b_rob, h, rate, gam, ntilesFull);
    }
    // 1b) scalar tail for N % 16 rows (not taken for N = 100000)
    if (tailRows > 0) {
        tail_gemm_kernel<<<8, 256, 0, stream>>>(
            x, W_fc, b_fc, W_rate, W_rob, b_rob, h, rate, gam,
            ntilesFull * 16, N);
    }
    // 2) edge gather + scatter-add
    {
        long long waves = E;
        int blocks = (int)((waves + 7) / 8);      // 8 waves per block
        if (blocks > 262144) blocks = 262144;     // grid-stride covers the rest
        edge_scatter_kernel<<<blocks, 256, 0, stream>>>(ei, h, partial, cnt, E);
    }
    // 3) combine + LayerNorm
    {
        int blocks = (N + 7) / 8;
        finalize_kernel<<<blocks, 256, 0, stream>>>(
            h, rate, gam, partial, cnt, degree, ln_g, ln_b, out, N);
    }
}
